// SimpleQuantMHA_14207751815652
// MI455X (gfx1250) — 8-bit matrix operand order repaired and checked by reading, compiled, not run on hardware
//
#include <hip/hip_runtime.h>
#include <hip/hip_bf16.h>
#include <stdint.h>


typedef int v8i __attribute__((ext_vector_type(8)));

#define S_LEN 2048
#define BATCH 2
#define EMB   1024
#define HEADS 16
#define MROWS (S_LEN * BATCH)
#define EPS_F 1e-8f

#define OFF_U    ((size_t)0)
#define OFF_F    ((size_t)256)
#define OFF_ROWM ((size_t)4096)
#define OFF_ROWZ ((size_t)266240)
#define OFF_XQ   ((size_t)1  << 20)
#define OFF_WQ8  ((size_t)5  << 20)
#define OFF_WK8  ((size_t)6  << 20)
#define OFF_WV8  ((size_t)7  << 20)
#define OFF_WO8  ((size_t)8  << 20)
#define OFF_ACC  ((size_t)9  << 20)
#define OFF_Q8   ((size_t)25 << 20)
#define OFF_K8   ((size_t)29 << 20)
#define OFF_VT8  ((size_t)33 << 20)
#define OFF_O8   ((size_t)37 << 20)


__device__ __forceinline__ v8i frag_a_i8(const signed char* base, int lane) {
  const int hl = lane >> 4, m = lane & 15;
  v8i a;
#pragma unroll
  for (int i = 0; i < 8; ++i) {
    const int kb = ((i >> 1) << 4) + (hl << 3) + ((i & 1) << 2);
    a[i] = *(const int*)(base + m * 64 + kb);
  }
  return a;
}
__device__ __forceinline__ v8i frag_b_i8(const signed char* base, int strideK, int lane) {
  const int hl = lane >> 4, n = lane & 15;
  v8i b;
#pragma unroll
  for (int i = 0; i < 8; ++i) {
    const int kb = ((i >> 1) << 4) + (hl << 3) + ((i & 1) << 2);
    b[i] = *(const int*)(base + (size_t)n * strideK + kb);
  }
  return b;
}

__global__ void k_init(unsigned* U) {
  if (threadIdx.x == 0 && blockIdx.x == 0) {
    for (int i = 0; i < 16; ++i) U[i] = 0u;
    U[8] = 0x7F800000u;
  }
}

__global__ void k_finalize(unsigned* U, float* F, int mode) {
  if (threadIdx.x || blockIdx.x) return;
  switch (mode) {
    case 0:
      for (int i = 0; i < 5; ++i)
        F[i] = fmaxf(__uint_as_float(U[i]) / 127.0f, EPS_F);
      break;
    case 1: F[5] = fmaxf(F[0] * F[1] * (float)U[5] * 0.125f / 127.0f, EPS_F); break;
    case 2: F[6] = fmaxf(F[0] * F[2] * (float)U[6] / 127.0f, EPS_F); break;
    case 3: F[7] = fmaxf(F[0] * F[3] * (float)U[7] / 127.0f, EPS_F); break;
    case 4: F[8] = fmaxf((1.0f / __uint_as_float(U[8])) / 255.0f, EPS_F); break;
    case 5: F[9] = fmaxf(F[8] * F[7] * (float)U[9] / 127.0f, EPS_F); break;
    case 6: F[10] = fmaxf(F[9] * F[4] * (float)U[10] / 127.0f, EPS_F); break;
  }
}

__global__ __launch_bounds__(256) void k_absmax(const float* __restrict__ x, int n,
                                                unsigned* __restrict__ slot) {
  float m = 0.0f;
  for (int i = blockIdx.x * blockDim.x + threadIdx.x; i < n; i += blockDim.x * gridDim.x)
    m = fmaxf(m, fabsf(x[i]));
#pragma unroll
  for (int sh = 16; sh >= 1; sh >>= 1) m = fmaxf(m, __shfl_xor(m, sh));
  if ((threadIdx.x & 31) == 0) atomicMax(slot, __float_as_uint(m));
}

__global__ __launch_bounds__(256) void k_quant_i8(const float* __restrict__ x,
                                                  signed char* __restrict__ y, int n,
                                                  const float* __restrict__ F, int slot,
                                                  int lo) {
  const float inv = 1.0f / F[slot];
  for (int i = blockIdx.x * blockDim.x + threadIdx.x; i < n; i += blockDim.x * gridDim.x) {
    int q = (int)rintf(x[i] * inv);
    y[i] = (signed char)min(max(q, lo), 127);
  }
}

__global__ __launch_bounds__(256) void k_gemm_i8(
    const signed char* __restrict__ A, const signed char* __restrict__ Bw,
    int* __restrict__ C, int M, int N, int K, unsigned* __restrict__ absSlot) {
  __shared__ __align__(16) signed char lA[64 * 64];
  __shared__ __align__(16) signed char lB[64 * 64];
  __shared__ int red[8];
  const int tid = threadIdx.x;
  const int wv = tid >> 5, lane = tid & 31;
  const int hl = lane >> 4, l16 = lane & 15;
  const int row0 = blockIdx.y * 64, col0 = blockIdx.x * 64;
  const int mi = (wv * 2) >> 2;
  const int ni0 = (wv * 2) & 3, ni1 = ni0 + 1;
  v8i acc0 = {}, acc1 = {};
  const int srow = tid >> 2, scol = (tid & 3) << 4;
  const size_t aRow = (size_t)(row0 + srow) * K + scol;
  const size_t bRow = (size_t)(col0 + srow) * K + scol;
  for (int kt = 0; kt < K; kt += 64) {
    *(uint4*)(lA + srow * 64 + scol) = *(const uint4*)(A + aRow + kt);
    *(uint4*)(lB + srow * 64 + scol) = *(const uint4*)(Bw + bRow + kt);
    if (kt + 64 < K) {
      __builtin_prefetch(A + aRow + kt + 64, 0, 1);
      __builtin_prefetch(Bw + bRow + kt + 64, 0, 1);
    }
    __syncthreads();
    v8i af  = frag_a_i8(lA + mi * 16 * 64, lane);
    v8i bf0 = frag_b_i8(lB + ni0 * 16 * 64, 64, lane);
    v8i bf1 = frag_b_i8(lB + ni1 * 16 * 64, 64, lane);
    acc0 = __builtin_amdgcn_wmma_i32_16x16x64_iu8(true, af, true, bf0, acc0, false, false);
    acc1 = __builtin_amdgcn_wmma_i32_16x16x64_iu8(true, af, true, bf1, acc1, false, false);
    __syncthreads();
  }
  int lmax = 0;
#pragma unroll
  for (int r = 0; r < 8; ++r) {
    const int mr = row0 + mi * 16 + r + (hl << 3);
    C[(size_t)mr * N + col0 + ni0 * 16 + l16] = acc0[r];
    C[(size_t)mr * N + col0 + ni1 * 16 + l16] = acc1[r];
    lmax = max(lmax, max(abs(acc0[r]), abs(acc1[r])));
  }
#pragma unroll
  for (int sh = 16; sh >= 1; sh >>= 1) lmax = max(lmax, __shfl_xor(lmax, sh));
  if (lane == 0) red[wv] = lmax;
  __syncthreads();
  if (tid == 0) {
    int g = 0;
    for (int i = 0; i < 8; ++i) g = max(g, red[i]);
    atomicMax(absSlot, (unsigned)g);
  }
}

__global__ __launch_bounds__(256) void k_quant_qkv(
    const int* __restrict__ acc, signed char* __restrict__ dst,
    const float* __restrict__ F, int wslot, int qslot, int mode) {
  const float pre = F[0] * F[wslot] * (mode == 0 ? 0.125f : 1.0f);
  const float inv = pre / F[qslot];
  for (int i = blockIdx.x * blockDim.x + threadIdx.x; i < MROWS * EMB;
       i += blockDim.x * gridDim.x) {
    const int row = i >> 10, col = i & 1023;
    const int s = row >> 1, bb = row & 1;
    const int h = col >> 6, d = col & 63;
    int q = (int)rintf((float)acc[i] * inv);
    q = min(max(q, -128), 127);
    const size_t bh = (size_t)bb * HEADS + h;
    const size_t off = (mode == 2) ? ((bh * 64 + d) * S_LEN + s)
                                   : ((bh * S_LEN + s) * 64 + d);
    dst[off] = (signed char)q;
  }
}

__global__ __launch_bounds__(256) void k_attn_stats(
    const signed char* __restrict__ q8, const signed char* __restrict__ k8,
    float* __restrict__ rowM, float* __restrict__ rowZ,
    unsigned* __restrict__ minZ, const float* __restrict__ F) {
  const int wv = threadIdx.x >> 5, lane = threadIdx.x & 31;
  const int task = blockIdx.x * 8 + wv;
  const int bh = task >> 7, mb = task & 127;
  const int hl = lane >> 4, l16 = lane & 15;
  const signed char* qb = q8 + ((size_t)bh * S_LEN + mb * 16) * 64;
  const signed char* kb = k8 + (size_t)bh * S_LEN * 64;
  const float sqk = F[5] * F[6];
  const v8i aq = frag_a_i8(qb, lane);
  float mx[8];
#pragma unroll
  for (int r = 0; r < 8; ++r) mx[r] = -3.0e38f;
  for (int t0 = 0; t0 < S_LEN; t0 += 16) {
    v8i bf = frag_b_i8(kb + (size_t)t0 * 64, 64, lane);
    v8i d = {};
    d = __builtin_amdgcn_wmma_i32_16x16x64_iu8(true, aq, true, bf, d, false, false);
#pragma unroll
    for (int r = 0; r < 8; ++r) mx[r] = fmaxf(mx[r], (float)d[r] * sqk);
  }
#pragma unroll
  for (int r = 0; r < 8; ++r)
#pragma unroll
    for (int sh = 8; sh >= 1; sh >>= 1) mx[r] = fmaxf(mx[r], __shfl_xor(mx[r], sh));
  float Z[8];
#pragma unroll
  for (int r = 0; r < 8; ++r) Z[r] = 0.0f;
  for (int t0 = 0; t0 < S_LEN; t0 += 16) {
    v8i bf = frag_b_i8(kb + (size_t)t0 * 64, 64, lane);
    v8i d = {};
    d = __builtin_amdgcn_wmma_i32_16x16x64_iu8(true, aq, true, bf, d, false, false);
#pragma unroll
    for (int r = 0; r < 8; ++r) Z[r] += __expf((float)d[r] * sqk - mx[r]);
  }
#pragma unroll
  for (int r = 0; r < 8; ++r)
#pragma unroll
    for (int sh = 8; sh >= 1; sh >>= 1) Z[r] += __shfl_xor(Z[r], sh);
  if (l16 == 0) {
#pragma unroll
    for (int r = 0; r < 8; ++r) {
      const int row = bh * S_LEN + mb * 16 + r + (hl << 3);
      rowM[row] = mx[r];
      rowZ[row] = Z[r];
      atomicMin(minZ, __float_as_uint(Z[r]));
    }
  }
}

__global__ __launch_bounds__(256) void k_attn_pv(
    const signed char* __restrict__ q8, const signed char* __restrict__ k8,
    const signed char* __restrict__ vT8, const float* __restrict__ rowM,
    const float* __restrict__ rowZ, int* __restrict__ oacc,
    unsigned* __restrict__ absSlot, const float* __restrict__ F) {
  __shared__ __align__(16) unsigned char abuf[8][16 * 64];
  const int wv = threadIdx.x >> 5, lane = threadIdx.x & 31;
  const int task = blockIdx.x * 8 + wv;
  const int bh = task >> 7, mb = task & 127;
  const int bb = bh >> 4, hh = bh & 15;
  const int hl = lane >> 4, l16 = lane & 15;
  const signed char* qb = q8 + ((size_t)bh * S_LEN + mb * 16) * 64;
  const signed char* kb = k8 + (size_t)bh * S_LEN * 64;
  const signed char* vb = vT8 + (size_t)bh * 64 * S_LEN;
  const float sqk = F[5] * F[6];
  const float inv_sa = 1.0f / F[8];
  float m8[8], zi8[8];
#pragma unroll
  for (int r = 0; r < 8; ++r) {
    const int row = bh * S_LEN + mb * 16 + r + (hl << 3);
    m8[r] = rowM[row];
    zi8[r] = 1.0f / rowZ[row];
  }
  const v8i aq = frag_a_i8(qb, lane);
  v8i oAcc[4] = {{}, {}, {}, {}};
  unsigned char* ab = &abuf[wv][0];
  for (int t0 = 0; t0 < S_LEN; t0 += 64) {
#pragma unroll
    for (int tt = 0; tt < 4; ++tt) {
      v8i bf = frag_b_i8(kb + (size_t)(t0 + tt * 16) * 64, 64, lane);
      v8i d = {};
      d = __builtin_amdgcn_wmma_i32_16x16x64_iu8(true, aq, true, bf, d, false, false);
#pragma unroll
      for (int r = 0; r < 8; ++r) {
        const float p = __expf((float)d[r] * sqk - m8[r]) * zi8[r];
        int u = (int)rintf(p * inv_sa);
        ab[(r + (hl << 3)) * 64 + tt * 16 + l16] = (unsigned char)min(max(u, 0), 255);
      }
    }
    v8i pa;
#pragma unroll
    for (int i = 0; i < 8; ++i) {
      const int kbyte = ((i >> 1) << 4) + (hl << 3) + ((i & 1) << 2);
      pa[i] = *(const int*)(ab + l16 * 64 + kbyte);
    }
#pragma unroll
    for (int nt = 0; nt < 4; ++nt) {
      v8i bv = frag_b_i8(vb + (size_t)nt * 16 * S_LEN + t0, S_LEN, lane);
      oAcc[nt] = __builtin_amdgcn_wmma_i32_16x16x64_iu8(false  , pa,
                                                        true  , bv,
                                                        oAcc[nt], false, false);
    }
  }
  int lmax = 0;
#pragma unroll
  for (int nt = 0; nt < 4; ++nt)
#pragma unroll
    for (int r = 0; r < 8; ++r) {
      const int s = mb * 16 + r + (hl << 3);
      const int dcol = nt * 16 + l16;
      oacc[(size_t)(s * BATCH + bb) * EMB + hh * 64 + dcol] = oAcc[nt][r];
      lmax = max(lmax, abs(oAcc[nt][r]));
    }
#pragma unroll
  for (int sh = 16; sh >= 1; sh >>= 1) lmax = max(lmax, __shfl_xor(lmax, sh));
  if (lane == 0) atomicMax(absSlot, (unsigned)lmax);
}

__global__ __launch_bounds__(256) void k_quant_o(const int* __restrict__ acc,
                                                 signed char* __restrict__ dst,
                                                 const float* __restrict__ F) {
  const float inv = F[8] * F[7] / F[9];
  for (int i = blockIdx.x * blockDim.x + threadIdx.x; i < MROWS * EMB;
       i += blockDim.x * gridDim.x) {
    int q = (int)rintf((float)acc[i] * inv);
    dst[i] = (signed char)min(max(q, -128), 127);
  }
}

__global__ __launch_bounds__(256) void k_writeout(const int* __restrict__ acc,
                                                  float* __restrict__ out,
                                                  const float* __restrict__ F) {
  const float pre = F[9] * F[4];
  const float sout = F[10];
  const float inv = pre / sout;
  for (int i = blockIdx.x * blockDim.x + threadIdx.x; i < MROWS * EMB;
       i += blockDim.x * gridDim.x) {
    int q = (int)rintf((float)acc[i] * inv);
    out[i] = (float)min(max(q, -128), 127) * sout;
  }
}

extern "C" void kernel_launch(void* const* d_in, const int* in_sizes, int n_in,
                              void* d_out, int out_size, void* d_ws, size_t ws_size,
                              hipStream_t stream) {
  (void)in_sizes; (void)n_in; (void)out_size; (void)ws_size;
  const float* x  = (const float*)d_in[0];
  const float* Wq = (const float*)d_in[1];
  const float* Wk = (const float*)d_in[2];
  const float* Wv = (const float*)d_in[3];
  const float* Wo = (const float*)d_in[4];
  float* out = (float*)d_out;

  uint8_t* ws = (uint8_t*)d_ws;
  unsigned* U = (unsigned*)(ws + OFF_U);
  float* F    = (float*)(ws + OFF_F);
  float* rowM = (float*)(ws + OFF_ROWM);
  float* rowZ = (float*)(ws + OFF_ROWZ);
  signed char* xq  = (signed char*)(ws + OFF_XQ);
  signed char* wq8 = (signed char*)(ws + OFF_WQ8);
  signed char* wk8 = (signed char*)(ws + OFF_WK8);
  signed char* wv8 = (signed char*)(ws + OFF_WV8);
  signed char* wo8 = (signed char*)(ws + OFF_WO8);
  int* acc         = (int*)(ws + OFF_ACC);
  signed char* q8  = (signed char*)(ws + OFF_Q8);
  signed char* k8  = (signed char*)(ws + OFF_K8);
  signed char* vT8 = (signed char*)(ws + OFF_VT8);
  signed char* o8  = (signed char*)(ws + OFF_O8);

  const int nx = MROWS * EMB;
  const int nw = EMB * EMB;
  const dim3 blk(256);
  const dim3 gEW(2048);
  const dim3 gGEMM(EMB / 64, MROWS / 64);
  const dim3 gATT(512);

  k_init<<<1, 1, 0, stream>>>(U);

  k_absmax<<<1024, blk, 0, stream>>>(x, nx, &U[0]);
  k_absmax<<<512, blk, 0, stream>>>(Wq, nw, &U[1]);
  k_absmax<<<512, blk, 0, stream>>>(Wk, nw, &U[2]);
  k_absmax<<<512, blk, 0, stream>>>(Wv, nw, &U[3]);
  k_absmax<<<512, blk, 0, stream>>>(Wo, nw, &U[4]);
  k_finalize<<<1, 1, 0, stream>>>(U, F, 0);

  k_quant_i8<<<gEW, blk, 0, stream>>>(x, xq, nx, F, 0, -128);
  k_quant_i8<<<gEW, blk, 0, stream>>>(Wq, wq8, nw, F, 1, -127);
  k_quant_i8<<<gEW, blk, 0, stream>>>(Wk, wk8, nw, F, 2, -127);
  k_quant_i8<<<gEW, blk, 0, stream>>>(Wv, wv8, nw, F, 3, -127);
  k_quant_i8<<<gEW, blk, 0, stream>>>(Wo, wo8, nw, F, 4, -127);

  k_gemm_i8<<<gGEMM, blk, 0, stream>>>(xq, wq8, acc, MROWS, EMB, EMB, &U[5]);
  k_finalize<<<1, 1, 0, stream>>>(U, F, 1);
  k_quant_qkv<<<gEW, blk, 0, stream>>>(acc, q8, F, 1, 5, 0);

  k_gemm_i8<<<gGEMM, blk, 0, stream>>>(xq, wk8, acc, MROWS, EMB, EMB, &U[6]);
  k_finalize<<<1, 1, 0, stream>>>(U, F, 2);
  k_quant_qkv<<<gEW, blk, 0, stream>>>(acc, k8, F, 2, 6, 1);

  k_gemm_i8<<<gGEMM, blk, 0, stream>>>(xq, wv8, acc, MROWS, EMB, EMB, &U[7]);
  k_finalize<<<1, 1, 0, stream>>>(U, F, 3);
  k_quant_qkv<<<gEW, blk, 0, stream>>>(acc, vT8, F, 3, 7, 2);

  k_attn_stats<<<gATT, blk, 0, stream>>>(q8, k8, rowM, rowZ, &U[8], F);
  k_finalize<<<1, 1, 0, stream>>>(U, F, 4);
  k_attn_pv<<<gATT, blk, 0, stream>>>(q8, k8, vT8, rowM, rowZ, acc, &U[9], F);
  k_finalize<<<1, 1, 0, stream>>>(U, F, 5);
  k_quant_o<<<gEW, blk, 0, stream>>>(acc, o8, F);

  k_gemm_i8<<<gGEMM, blk, 0, stream>>>(o8, wo8, acc, MROWS, EMB, EMB, &U[10]);
  k_finalize<<<1, 1, 0, stream>>>(U, F, 6);
  k_writeout<<<gEW, blk, 0, stream>>>(acc, out, F);
}
